// MultiHeadAttention_9131100471662
// MI455X (gfx1250) — matrix operand order repaired and checked by reading, not compiled, not run on hardware
//
#include <hip/hip_runtime.h>
#include <hip/hip_bf16.h>

typedef unsigned short u16;
typedef unsigned int u32;
typedef __attribute__((ext_vector_type(16))) __bf16 v16bf;
typedef __attribute__((ext_vector_type(8)))  __bf16 v8bf;
typedef __attribute__((ext_vector_type(8)))  float  v8f;
typedef int v4i __attribute__((__vector_size__(4 * sizeof(int))));

#define BATCH 16
#define MTOK  512
#define HID   1024
#define HEADS 16
#define DH    64
#define MEMT  2
#define PAD   576            // 514 padded to a multiple of 64
#define NEGV  (-1.0e9f)

// Async global->LDS staging if the toolchain exposes the gfx1250 builtins.
#if __has_builtin(__builtin_amdgcn_global_load_async_to_lds_b128) && \
    __has_builtin(__builtin_amdgcn_s_wait_asynccnt)
#define HAVE_ASYNC_LDS 1
#else
#define HAVE_ASYNC_LDS 0
#endif

// ---------------- helpers ----------------

__device__ inline u16 f2bf(float f) {
  union { float f; u32 u; } x; x.f = f;
  u32 r = x.u + 0x7FFFu + ((x.u >> 16) & 1u);   // round-to-nearest-even
  return (u16)(r >> 16);
}

__device__ inline v8f vzero8() {
  v8f z;
#pragma unroll
  for (int i = 0; i < 8; ++i) z[i] = 0.0f;
  return z;
}

union FragU { v16bf v; v8bf h[2]; };

// A fragment 16x32 bf16: lane holds row (L&15); K = 8g+[0..7] and 16+8g+[0..7]
__device__ inline v16bf load_a_frag(const u16* rowptr, int g) {
  FragU f;
  f.h[0] = *(const v8bf*)(rowptr + 8 * g);
  f.h[1] = *(const v8bf*)(rowptr + 16 + 8 * g);
  return f.v;
}
// B fragment 32x16 bf16 (stored transposed: row = output column, contiguous K):
// lane holds column (L&15); K = 8g+[0..7] and 16+8g+[0..7]
__device__ inline v16bf load_b_frag(const u16* rowptr, int g) {
  FragU f;
  f.h[0] = *(const v8bf*)(rowptr + 8 * g);
  f.h[1] = *(const v8bf*)(rowptr + 16 + 8 * g);
  return f.v;
}

__device__ inline v8f wmma_bf16(v16bf a, v16bf b, v8f c) {
  return __builtin_amdgcn_wmma_f32_16x16x32_bf16(false, a, false, b, (short)0, c, false, false);
}

#if HAVE_ASYNC_LDS
__device__ inline void async_copy16(const u16* g, u16* l) {
  __builtin_amdgcn_global_load_async_to_lds_b128(
      (__attribute__((address_space(1))) v4i*)g,
      (__attribute__((address_space(3))) v4i*)l, 0, 0);
}
#endif

// ---------------- pack / convert ----------------

// Build memory-prepended, zero-padded bf16 copy (row-major) AND its transpose.
__global__ void k_pack(const float* __restrict__ src, const float* __restrict__ mem,
                       u16* __restrict__ dstR, u16* __restrict__ dstT) {
  size_t idx = (size_t)blockIdx.x * 256 + threadIdx.x;
  if (idx >= (size_t)BATCH * PAD * HID) return;
  int k = (int)(idx % HID);
  size_t t = idx / HID;
  int m = (int)(t % PAD);
  int b = (int)(t / PAD);
  float v = 0.0f;
  if (m < MEMT)             v = mem[(size_t)m * HID + k];
  else if (m < MEMT + MTOK) v = src[((size_t)b * MTOK + (m - MEMT)) * HID + k];
  u16 h = f2bf(v);
  dstR[idx] = h;
  dstT[((size_t)b * HID + k) * PAD + m] = h;
}

__global__ void k_cvt(const float* __restrict__ src, u16* __restrict__ dst, int n) {
  int i = blockIdx.x * 256 + threadIdx.x;
  if (i < n) dst[i] = f2bf(src[i]);
}

// token flags: 1 = valid, 0 = masked (-1e9), -1 = padding (excluded from softmax)
__global__ void k_flags(const int* __restrict__ mask, float* __restrict__ flag) {
  int i = blockIdx.x * 256 + threadIdx.x;
  if (i >= BATCH * PAD) return;
  int p = i % PAD, b = i / PAD;
  float f;
  if (p < MEMT)             f = 1.0f;
  else if (p < MEMT + MTOK) f = (mask[b * MTOK + (p - MEMT)] != 0) ? 1.0f : 0.0f;
  else                      f = -1.0f;
  flag[i] = f;
}

// ---------------- generic bf16 WMMA GEMM ----------------
// C[b,m,n] = sum_k A[b, a_row0+m, k] * BT[b, n, k]    (BT = B transposed)
// 64x64 block, 4 waves; each wave: one 16-row tile x four 16-col tiles.
__global__ __launch_bounds__(128) void k_gemm(
    const u16* __restrict__ A, int a_row0, int lda, size_t a_bstride,
    const u16* __restrict__ BT, int ldb, size_t bt_bstride,
    float* __restrict__ Cf, u16* __restrict__ Cb, int ldc, size_t c_bstride,
    int Ksize) {
  __shared__ u16 As[64 * 40];
  __shared__ u16 Bs[64 * 40];
  int b = blockIdx.z;
  int m0 = blockIdx.x * 64, n0 = blockIdx.y * 64;
  int t = threadIdx.x;
  int wave = t >> 5, L = t & 31, r = L & 15, g = (L >> 4) & 1;
  const u16* Ab = A + b * a_bstride + (size_t)a_row0 * lda;
  const u16* Bb = BT + b * bt_bstride;
  v8f acc[4];
#pragma unroll
  for (int nt = 0; nt < 4; ++nt) acc[nt] = vzero8();

  for (int kc = 0; kc < Ksize; kc += 32) {
    __syncthreads();
#pragma unroll
    for (int i = 0; i < 2; ++i) {
      int c = t * 2 + i;
      int row = c >> 2, ko = (c & 3) * 8;
      const u16* ga = Ab + (size_t)(m0 + row) * lda + kc + ko;
      const u16* gb = Bb + (size_t)(n0 + row) * ldb + kc + ko;
#if HAVE_ASYNC_LDS
      async_copy16(ga, &As[row * 40 + ko]);   // global_load_async_to_lds_b128
      async_copy16(gb, &Bs[row * 40 + ko]);
#else
      *(uint4*)(&As[row * 40 + ko]) = *(const uint4*)ga;
      *(uint4*)(&Bs[row * 40 + ko]) = *(const uint4*)gb;
#endif
      if (kc + 32 < Ksize) {  // global_prefetch_b8 of next K tile
        __builtin_prefetch(ga + 32, 0, 1);
        __builtin_prefetch(gb + 32, 0, 1);
      }
    }
#if HAVE_ASYNC_LDS
    __builtin_amdgcn_s_wait_asynccnt(0);
#endif
    __syncthreads();
    v16bf af = load_a_frag(&As[(wave * 16 + r) * 40], g);
#pragma unroll
    for (int nt = 0; nt < 4; ++nt) {
      v16bf bf = load_b_frag(&Bs[(nt * 16 + r) * 40], g);
      acc[nt] = wmma_bf16(af, bf, acc[nt]);
    }
  }
#pragma unroll
  for (int nt = 0; nt < 4; ++nt) {
#pragma unroll
    for (int v = 0; v < 8; ++v) {
      int col = n0 + nt * 16 + r;
      int row = m0 + wave * 16 + 8 * g + v;
      size_t off = b * c_bstride + (size_t)row * ldc + col;
      if (Cb) Cb[off] = f2bf(acc[nt][v]);
      else    Cf[off] = acc[nt][v];
    }
  }
}

// ---------------- softmax statistics (online, streaming over columns) ----------------
// aff(m,n) = sum_d Qp[b,m,h*64+d] * Kp[b,n,h*64+d]; masked per flags; row max/sum.
__global__ __launch_bounds__(32) void k_stats(
    const u16* __restrict__ Qp, const u16* __restrict__ Kp,
    const float* __restrict__ flagQ, const float* __restrict__ flagK,
    float* __restrict__ omax, float* __restrict__ osum) {
  const float INF = __builtin_inff();
  int b = blockIdx.z, h = blockIdx.y, m0 = blockIdx.x * 16;
  int L = threadIdx.x, r = L & 15, g = (L >> 4) & 1;

  const u16* arow = Qp + ((size_t)b * PAD + m0 + r) * HID + h * DH;
  v16bf a0 = load_a_frag(arow, g);
  v16bf a1 = load_a_frag(arow + 32, g);

  float fq[8], rmax[8], rsum[8];
#pragma unroll
  for (int v = 0; v < 8; ++v) {
    fq[v] = flagQ[(size_t)b * PAD + m0 + 8 * g + v];
    rmax[v] = -INF; rsum[v] = 0.0f;
  }

  for (int n0 = 0; n0 < PAD; n0 += 16) {
    const u16* brow = Kp + ((size_t)b * PAD + n0 + r) * HID + h * DH;
    v16bf b0 = load_b_frag(brow, g);
    v16bf b1 = load_b_frag(brow + 32, g);
    v8f c = vzero8();
    c = wmma_bf16(a0, b0, c);
    c = wmma_bf16(a1, b1, c);
    float fk = flagK[(size_t)b * PAD + n0 + r];
#pragma unroll
    for (int v = 0; v < 8; ++v) {
      float a = c[v];
      a = (fq[v] < 0.0f || fk < 0.0f) ? -INF
          : ((fq[v] < 0.5f || fk < 0.5f) ? NEGV : a);
      float tmax = a;
#pragma unroll
      for (int off = 1; off < 16; off <<= 1)
        tmax = fmaxf(tmax, __shfl_xor(tmax, off, 16));
      float nm = fmaxf(rmax[v], tmax);
      if (nm > -INF) {
        float p = __expf(a - nm);      // a == -INF -> 0 (excluded)
#pragma unroll
        for (int off = 1; off < 16; off <<= 1)
          p += __shfl_xor(p, off, 16);
        rsum[v] = rsum[v] * __expf(rmax[v] - nm) + p;
        rmax[v] = nm;
      }
    }
  }
  if (r == 0) {
#pragma unroll
    for (int v = 0; v < 8; ++v) {
      float mx = rmax[v], sm = rsum[v];
      if (!(sm > 0.0f)) { mx = 0.0f; sm = 1.0f; }   // fully-padded row
      size_t o = ((size_t)(b * HEADS + h)) * PAD + m0 + 8 * g + v;
      omax[o] = mx; osum[o] = sm;
    }
  }
}

// ---------------- head-mean probabilities ----------------
// attn[b,m,n] = (1/H) * sum_h exp(aff_h(m,n) - rmax)/rsum, written as bf16.
__global__ __launch_bounds__(32) void k_attn(
    const u16* __restrict__ Qp, const u16* __restrict__ Kp,
    const float* __restrict__ flagQ, const float* __restrict__ flagK,
    const float* __restrict__ smax, const float* __restrict__ ssum,
    u16* __restrict__ attn) {
  int b = blockIdx.z, m0 = blockIdx.y * 16, n0 = blockIdx.x * 16;
  int L = threadIdx.x, r = L & 15, g = (L >> 4) & 1;

  float acc[8], fq[8];
#pragma unroll
  for (int v = 0; v < 8; ++v) {
    acc[v] = 0.0f;
    fq[v] = flagQ[(size_t)b * PAD + m0 + 8 * g + v];
  }
  float fk = flagK[(size_t)b * PAD + n0 + r];

  for (int h = 0; h < HEADS; ++h) {
    const u16* arow = Qp + ((size_t)b * PAD + m0 + r) * HID + h * DH;
    const u16* brow = Kp + ((size_t)b * PAD + n0 + r) * HID + h * DH;
    v16bf a0 = load_a_frag(arow, g);
    v16bf a1 = load_a_frag(arow + 32, g);
    v16bf b0 = load_b_frag(brow, g);
    v16bf b1 = load_b_frag(brow + 32, g);
    v8f c = vzero8();
    c = wmma_bf16(a0, b0, c);
    c = wmma_bf16(a1, b1, c);
#pragma unroll
    for (int v = 0; v < 8; ++v) {
      size_t so = ((size_t)(b * HEADS + h)) * PAD + m0 + 8 * g + v;
      float p;
      if (fq[v] < 0.0f || fk < 0.0f) {
        p = 0.0f;
      } else {
        float a = (fq[v] < 0.5f || fk < 0.5f) ? NEGV : c[v];
        p = __expf(a - smax[so]) * (1.0f / ssum[so]);
      }
      acc[v] += p;
    }
  }
#pragma unroll
  for (int v = 0; v < 8; ++v)
    attn[((size_t)b * PAD + m0 + 8 * g + v) * PAD + n0 + r] =
        f2bf(acc[v] * (1.0f / HEADS));
}

// ---------------- host side ----------------

extern "C" void kernel_launch(void* const* d_in, const int* in_sizes, int n_in,
                              void* d_out, int out_size, void* d_ws, size_t ws_size,
                              hipStream_t stream) {
  const float* x     = (const float*)d_in[0];
  const float* y     = (const float*)d_in[1];
  const int*   mx    = (const int*)d_in[2];
  const int*   my    = (const int*)d_in[3];
  const float* Wx    = (const float*)d_in[4];
  const float* Wy    = (const float*)d_in[5];
  const float* xmem  = (const float*)d_in[6];
  const float* ymem  = (const float*)d_in[7];

  char* ws = (char*)d_ws;
  size_t off = 0;
  auto alloc = [&](size_t bytes) -> size_t {
    size_t r = off; off = (off + bytes + 255) & ~(size_t)255; return r;
  };
  const size_t PACK_ELEMS = (size_t)BATCH * PAD * HID;
  size_t oXm   = alloc(PACK_ELEMS * 2);
  size_t oYm   = alloc(PACK_ELEMS * 2);
  size_t oXmT  = alloc(PACK_ELEMS * 2);
  size_t oYmT  = alloc(PACK_ELEMS * 2);
  size_t oWx   = alloc((size_t)HID * HID * 2);
  size_t oWy   = alloc((size_t)HID * HID * 2);
  size_t oXp   = alloc(PACK_ELEMS * 2);
  size_t oYp   = alloc(PACK_ELEMS * 2);
  size_t oFx   = alloc((size_t)BATCH * PAD * 4);
  size_t oFy   = alloc((size_t)BATCH * PAD * 4);
  size_t oRmx  = alloc((size_t)BATCH * HEADS * PAD * 4);   // row stats (attn_Y)
  size_t oRsm  = alloc((size_t)BATCH * HEADS * PAD * 4);
  size_t oCmx  = alloc((size_t)BATCH * HEADS * PAD * 4);   // col stats (attn_X, via transpose)
  size_t oCsm  = alloc((size_t)BATCH * HEADS * PAD * 4);
  size_t oAY   = alloc((size_t)BATCH * PAD * PAD * 2);     // attn_Y   [b][m][n] bf16
  size_t oAXT  = alloc((size_t)BATCH * PAD * PAD * 2);     // attn_X^T [b][n][m] bf16

  u16* Xm  = (u16*)(ws + oXm);   u16* Ym  = (u16*)(ws + oYm);
  u16* XmT = (u16*)(ws + oXmT);  u16* YmT = (u16*)(ws + oYmT);
  u16* Wxb = (u16*)(ws + oWx);   u16* Wyb = (u16*)(ws + oWy);
  u16* Xp  = (u16*)(ws + oXp);   u16* Yp  = (u16*)(ws + oYp);
  float* Fx = (float*)(ws + oFx);  float* Fy = (float*)(ws + oFy);
  float* Rmx = (float*)(ws + oRmx); float* Rsm = (float*)(ws + oRsm);
  float* Cmx = (float*)(ws + oCmx); float* Csm = (float*)(ws + oCsm);
  u16* AY  = (u16*)(ws + oAY);   u16* AXT = (u16*)(ws + oAXT);

  // 1. pack + transpose + bf16-convert inputs / weights / flags
  int packBlocks = (int)((PACK_ELEMS + 255) / 256);
  k_pack<<<packBlocks, 256, 0, stream>>>(x, xmem, Xm, XmT);
  k_pack<<<packBlocks, 256, 0, stream>>>(y, ymem, Ym, YmT);
  k_cvt<<<(HID * HID + 255) / 256, 256, 0, stream>>>(Wx, Wxb, HID * HID);
  k_cvt<<<(HID * HID + 255) / 256, 256, 0, stream>>>(Wy, Wyb, HID * HID);
  k_flags<<<(BATCH * PAD + 255) / 256, 256, 0, stream>>>(mx, Fx);
  k_flags<<<(BATCH * PAD + 255) / 256, 256, 0, stream>>>(my, Fy);

  // 2. projections: Xp = Xm @ Wx^T  (Wx rows are already the transposed-B layout)
  {
    dim3 g(PAD / 64, HID / 64, BATCH);
    k_gemm<<<g, 128, 0, stream>>>(Xm, 0, HID, (size_t)PAD * HID,
                                  Wxb, HID, 0,
                                  nullptr, Xp, HID, (size_t)PAD * HID, HID);
    k_gemm<<<g, 128, 0, stream>>>(Ym, 0, HID, (size_t)PAD * HID,
                                  Wyb, HID, 0,
                                  nullptr, Yp, HID, (size_t)PAD * HID, HID);
  }

  // 3. softmax statistics: row-stats of aff (attn_Y) and of aff^T (col-stats -> attn_X)
  {
    dim3 g(PAD / 16, HEADS, BATCH);
    k_stats<<<g, 32, 0, stream>>>(Xp, Yp, Fx, Fy, Rmx, Rsm);
    k_stats<<<g, 32, 0, stream>>>(Yp, Xp, Fy, Fx, Cmx, Csm);
  }

  // 4. head-mean probabilities (recompute affinity via WMMA)
  {
    dim3 g(PAD / 16, PAD / 16, BATCH);
    k_attn<<<g, 32, 0, stream>>>(Xp, Yp, Fx, Fy, Rmx, Rsm, AY);   // attn_Y [m][n]
    k_attn<<<g, 32, 0, stream>>>(Yp, Xp, Fy, Fx, Cmx, Csm, AXT);  // attn_X^T [n][m]
  }

  // 5. outputs: X_in_Y = attn_X^T @ Xmem ; Y_in_X = attn_Y @ Ymem  (slice rows MEM..MEM+512)
  {
    float* outX = (float*)d_out;
    float* outY = outX + (size_t)BATCH * MTOK * HID;
    dim3 g(MTOK / 64, HID / 64, BATCH);
    k_gemm<<<g, 128, 0, stream>>>(AXT, MEMT, PAD, (size_t)PAD * PAD,
                                  XmT, PAD, (size_t)HID * PAD,
                                  outX, nullptr, HID, (size_t)MTOK * HID, PAD);
    k_gemm<<<g, 128, 0, stream>>>(AY, MEMT, PAD, (size_t)PAD * PAD,
                                  YmT, PAD, (size_t)HID * PAD,
                                  outY, nullptr, HID, (size_t)MTOK * HID, PAD);
  }
}
